// ContrastiveLoss_46677704573490
// MI455X (gfx1250) — compile-verified, matrix operand order checked by reading, not run on hardware
//
#include <hip/hip_runtime.h>

typedef __attribute__((ext_vector_type(16))) _Float16 v16h;
typedef __attribute__((ext_vector_type(8)))  _Float16 v8h;
typedef __attribute__((ext_vector_type(8)))  float    v8f;

#define NROW 8192
#define KDIM 1024
#define INV_T 14.285714285714286f       // 1/0.07
#define K1LOG 20.609937685963484f       // (1/0.07) * log2(e)
#define EPS_N 1e-8f

// ---------------------------------------------------------------------------
// Kernel 1: row-wise L2 normalize fp32 -> fp16 (WMMA operand precision)
// ---------------------------------------------------------------------------
__global__ void __launch_bounds__(256) cl_normalize(const float* __restrict__ z,
                                                    _Float16* __restrict__ zn) {
    __shared__ float red[256];
    const int row = blockIdx.x;
    const int tid = threadIdx.x;
    const float* zr = z + (size_t)row * KDIM;

    float v[4];
    float ss = 0.f;
#pragma unroll
    for (int i = 0; i < 4; ++i) {
        v[i] = zr[tid + 256 * i];
        ss += v[i] * v[i];
    }
    red[tid] = ss;
    __syncthreads();
    for (int s = 128; s > 0; s >>= 1) {
        if (tid < s) red[tid] += red[tid + s];
        __syncthreads();
    }
    const float inv = 1.f / fmaxf(sqrtf(red[0]), EPS_N);
    _Float16* zo = zn + (size_t)row * KDIM;
#pragma unroll
    for (int i = 0; i < 4; ++i) zo[tid + 256 * i] = (_Float16)(v[i] * inv);
}

// ---------------------------------------------------------------------------
// Kernel 2: fused  zn @ zn^T  (WMMA f16 -> f32)  + mask/temp + partial lse.
// 64-row stripes, four 16-row sub-stripes -> each B fragment feeds 4 wmmas
// (total L2 B-traffic 2 GB).  2 WGs per stripe (half column sweeps) keep the
// grid at 256 for a ~256-WGP part.  Stripe rows are INTERLEAVED in LDS
// (ldsRow = r16*4 + s) so a lane's 4 sub-stripe fragments sit within 8 KB and
// every ds_load offset is an immediate.  Diagonal-mask and positive-pair
// handling are hoisted behind a wave-uniform scalar branch (each fires for at
// most 1 of a wave's 32 tiles); the hot path is branch-free
// acc += exp2(fma(c, K1, -K1)).  Fixed shift (cos/T <= 1/T) -> no online max;
// fixed-order reductions -> deterministic.
// ---------------------------------------------------------------------------
__global__ void __launch_bounds__(256) cl_sim_lse(const _Float16* __restrict__ zn,
                                                  float* __restrict__ expSum,  // [2][NROW]
                                                  float* __restrict__ posG) {  // [NROW]
    __shared__ _Float16 Atile[64 * KDIM];      // 128 KB: 64 rows, interleaved
    __shared__ float    part[64 * 16 * 8];     // 32 KB: [row64][col16][wave]
    __shared__ float    posv[64];              // positive-pair logits

    const int tid    = threadIdx.x;
    const int wave   = tid >> 5;               // wave32 (gfx1250 is wave32-only)
    const int lane   = tid & 31;
    const int stripe = blockIdx.x >> 1;
    const int half   = blockIdx.x & 1;         // which half of the column sweep
    const int m0     = stripe << 6;            // 64-row stripe base

    // Stage the 64-row A stripe into LDS with row interleave: global row
    // g = s*16 + r16  ->  LDS row r16*4 + s.
    {
        const v8h* src = (const v8h*)(zn + (size_t)m0 * KDIM);
        v8h*       dst = (v8h*)Atile;
#pragma unroll
        for (int i = 0; i < 32; ++i) {
            const int c    = tid + 256 * i;    // v8h chunk 0..8191
            const int g    = c >> 7;           // stripe row 0..63
            const int cc   = c & 127;          // column chunk
            const int lrow = ((g & 15) << 2) | (g >> 4);
            dst[(lrow << 7) | cc] = src[c];
        }
    }
    __syncthreads();

    // Per-lane fragment addressing (identical math for A and B operands):
    //   halves [kb + 8*(lane>=16), +8) and [kb + 16 + 8*(lane>=16), +8)
    const int r16 = lane & 15;                 // A matrix row / B column
    const int mh  = (lane >> 4) << 3;          // 0 or 8 (K offset; also C row offset)

    float acc[4][8];
#pragma unroll
    for (int s = 0; s < 4; ++s)
#pragma unroll
        for (int v = 0; v < 8; ++v) acc[s][v] = 0.f;

    // Lane base into interleaved Atile; sub-stripe s is +s*KDIM halves (2 KB).
    const _Float16* abase = Atile + (size_t)(r16 << 2) * KDIM + mh;

    const int diagBase = m0 >> 4;              // 4 tiles containing the diagonal
    const int posBase  = (m0 ^ 4096) >> 4;     // 4 tiles containing positives

    const int ntEnd = (half + 1) << 8;         // 256 tiles per half
    for (int nt = (half << 8) + wave; nt < ntEnd; nt += 8) {
        const int n0 = nt << 4;
        const _Float16* brow = zn + (size_t)(n0 + r16) * KDIM + mh;

        v8f c[4];
#pragma unroll
        for (int s = 0; s < 4; ++s) c[s] = (v8f){};

#pragma clang loop unroll_count(2)
        for (int kb = 0; kb < KDIM; kb += 32) {
            __builtin_prefetch(brow + kb + 256, 0, 3);    // global_prefetch_b8
            union { v16h v; v8h h[2]; } b;
            b.h[0] = *(const v8h*)(brow + kb);             // global_load_b128 (L2 hit)
            b.h[1] = *(const v8h*)(brow + kb + 16);
#pragma unroll
            for (int s = 0; s < 4; ++s) {
                union { v16h v; v8h h[2]; } a;             // ds_load_b128, imm offsets
                a.h[0] = *(const v8h*)(abase + s * KDIM + kb);
                a.h[1] = *(const v8h*)(abase + s * KDIM + kb + 16);
                c[s] = __builtin_amdgcn_wmma_f32_16x16x32_f16(
                           false, a.v, false, b.v, (short)0, c[s], false, false);
            }
        }

        // Epilogue. nt is wave-uniform: force a scalar branch so the rare
        // diag/pos tiles don't tax the common path with exec-mask churn.
        const int ntu = __builtin_amdgcn_readfirstlane(nt);
        if ((unsigned)(ntu - diagBase) < 4u || (unsigned)(ntu - posBase) < 4u) {
            const int gcol = n0 + r16;
#pragma unroll
            for (int s = 0; s < 4; ++s)
#pragma unroll
                for (int v = 0; v < 8; ++v) {
                    const int m16  = (s << 4) + v + mh;
                    const int grow = m0 + m16;
                    if (grow != gcol)
                        acc[s][v] += exp2f(fmaf(c[s][v], K1LOG, -K1LOG));
                    if (gcol == (grow ^ 4096)) posv[m16] = c[s][v] * INV_T;
                }
        } else {
#pragma unroll
            for (int s = 0; s < 4; ++s)
#pragma unroll
                for (int v = 0; v < 8; ++v)
                    acc[s][v] += exp2f(fmaf(c[s][v], K1LOG, -K1LOG));
        }
    }

    // Deterministic cross-lane/cross-wave reduction via fixed-order LDS sums.
#pragma unroll
    for (int s = 0; s < 4; ++s)
#pragma unroll
        for (int v = 0; v < 8; ++v) {
            const int m16 = (s << 4) + v + mh;
            part[(m16 << 7) | (r16 << 3) | wave] = acc[s][v];
        }
    __syncthreads();

    if (tid < 64) {
        const float* p = part + (tid << 7);
        float sum = 0.f;
#pragma unroll 8
        for (int i = 0; i < 128; ++i) sum += p[i];
        expSum[half * NROW + m0 + tid] = sum;
        // This half owns the positive block iff row^4096 lands in its columns.
        const bool ownPos = (m0 < 4096) ? (half == 1) : (half == 0);
        if (ownPos) posG[m0 + tid] = posv[tid];
    }
}

// ---------------------------------------------------------------------------
// Kernel 3: combine half-sweep partials, per-row NLL, deterministic mean.
// ---------------------------------------------------------------------------
__global__ void __launch_bounds__(256) cl_nll_mean(const float* __restrict__ expSum,
                                                   const float* __restrict__ posG,
                                                   float* __restrict__ out) {
    __shared__ float red[256];
    const int tid = threadIdx.x;
    float s = 0.f;
    for (int i = tid; i < NROW; i += 256) {
        const float tot = expSum[i] + expSum[NROW + i];
        s += -posG[i] + logf(tot) + INV_T;   // nll_i = -pos + logsumexp
    }
    red[tid] = s;
    __syncthreads();
    for (int k = 128; k > 0; k >>= 1) {
        if (tid < k) red[tid] += red[tid + k];
        __syncthreads();
    }
    if (tid == 0) out[0] = red[0] / (float)NROW;
}

// ---------------------------------------------------------------------------
extern "C" void kernel_launch(void* const* d_in, const int* in_sizes, int n_in,
                              void* d_out, int out_size, void* d_ws, size_t ws_size,
                              hipStream_t stream) {
    (void)in_sizes; (void)n_in; (void)out_size; (void)ws_size;
    const float* z = (const float*)d_in[0];

    char* ws = (char*)d_ws;
    _Float16* zn     = (_Float16*)ws;                                  // 16 MB
    float*    expSum = (float*)(ws + (size_t)NROW * KDIM * 2);         // 64 KB
    float*    posG   = (float*)(ws + (size_t)NROW * KDIM * 2 + 2 * NROW * sizeof(float));
    float*    out    = (float*)d_out;

    cl_normalize<<<NROW, 256, 0, stream>>>(z, zn);
    cl_sim_lse<<<(NROW / 64) * 2, 256, 0, stream>>>(zn, expSum, posG);
    cl_nll_mean<<<1, 256, 0, stream>>>(expSum, posG, out);
}
